// ODENet_12987981103399
// MI455X (gfx1250) — matrix operand order repaired and checked by reading, not compiled, not run on hardware
//
#include <hip/hip_runtime.h>

// ---------------------------------------------------------------------------
// Augmented Neural ODE (IN=128, H=256) integrated with fixed-step RK4 (32
// steps => 128 MLP evals), all-WMMA f16 compute, transposed (feature x batch)
// formulation so layer chaining needs only per-lane f16 packing (no lane exchange).
// Weights pre-packed to CDNA5 A-fragment layout in d_ws, staged to LDS.
// A compiler memory barrier per RK4 stage stops LICM from hoisting the
// loop-invariant LDS weight loads into (spilled) registers.
// ---------------------------------------------------------------------------

typedef __attribute__((ext_vector_type(16))) _Float16 v16h;
typedef __attribute__((ext_vector_type(8)))  float    v8f;

#define NB_ROWS 65536
#define NDATA   118
#define SDIM    128
#define HDIM    256
#define ODIM    10
#define NSTEPS  32

// LDS layout (bytes): packed f16 weights, then f32 biases / head
#define LDS_B1    262144                   // 256 f32
#define LDS_B2    (262144 + 1024)          // 256 f32
#define LDS_B3    (262144 + 2048)          // 128 f32
#define LDS_WL    (262144 + 2560)          // 1280 f32 (128x10)
#define LDS_BL    (262144 + 2560 + 5120)   // 10 f32
#define LDS_TOTAL (262144 + 2560 + 5120 + 64)

// ---------------------------------------------------------------------------
// Prep kernel: pack W1,W2,W3 (f32 row-major [in][out]) into f16 A-fragments
// of Wt (out x in).  A 16-bit 16x32 fragment layout (ISA 7.12.2):
//   lane 0-15 : row m = 16t+lane,  halves j -> K = 32c + (j<8 ? j : j+8)
//   lane16-31 : row m = 16t+lane-16, halves j -> K = 32c + 8 + (j<8 ? j : j+8)
// Fragment f stored at wp + f*512 halves, lane-major (32B per lane).
// Fragment order: W1 frags 0..63 (t*4+c), W2 64..191 (t*8+c), W3 192..255.
// ---------------------------------------------------------------------------
__global__ void prep_pack_weights(const float* __restrict__ W1,
                                  const float* __restrict__ W2,
                                  const float* __restrict__ W3,
                                  unsigned short* __restrict__ wp) {
  int gid  = blockIdx.x * blockDim.x + threadIdx.x;   // 8192 threads
  int frag = gid >> 5;
  int lane = gid & 31;
  if (frag >= 256) return;

  const float* W;
  int t, c, fanOut;
  if (frag < 64)       { W = W1; t = frag >> 2;         c = frag & 3;         fanOut = 256; }
  else if (frag < 192) { W = W2; t = (frag - 64) >> 3;  c = (frag - 64) & 7;  fanOut = 256; }
  else                 { W = W3; t = (frag - 192) >> 3; c = (frag - 192) & 7; fanOut = 128; }

  int m     = 16 * t + (lane & 15);
  int kbase = 32 * c + ((lane >> 4) << 3);

  unsigned u[8];
#pragma unroll
  for (int i = 0; i < 8; ++i) {
    int j0 = 2 * i, j1 = 2 * i + 1;
    int k0 = kbase + (j0 < 8 ? j0 : j0 + 8);
    int k1 = kbase + (j1 < 8 ? j1 : j1 + 8);
    _Float16 h0 = (_Float16)W[k0 * fanOut + m];
    _Float16 h1 = (_Float16)W[k1 * fanOut + m];
    unsigned short s0 = __builtin_bit_cast(unsigned short, h0);
    unsigned short s1 = __builtin_bit_cast(unsigned short, h1);
    u[i] = (unsigned)s0 | ((unsigned)s1 << 16);
  }
  uint4* dst = (uint4*)(wp + (size_t)frag * 512 + lane * 16);
  dst[0] = make_uint4(u[0], u[1], u[2], u[3]);
  dst[1] = make_uint4(u[4], u[5], u[6], u[7]);
}

// ---------------------------------------------------------------------------
// helpers
// ---------------------------------------------------------------------------
static __device__ inline unsigned pkf16(float a, float b) {
  _Float16 ha = (_Float16)a, hb = (_Float16)b;
  unsigned short sa = __builtin_bit_cast(unsigned short, ha);
  unsigned short sb = __builtin_bit_cast(unsigned short, hb);
  return (unsigned)sa | ((unsigned)sb << 16);
}

// Convert a pair of D-layout f32 tiles (feat tiles 2c, 2c+1) into one f16
// B-fragment (K = 8*hi + (j<8 ? j : j+8) layout).  No exchange with the
// partner lane is needed.
static __device__ inline v16h packpair(v8f a, v8f b, int hi) {
  unsigned pa[4], pb[4];
#pragma unroll
  for (int i = 0; i < 4; ++i) {
    pa[i] = pkf16(a[2 * i], a[2 * i + 1]);
    pb[i] = pkf16(b[2 * i], b[2 * i + 1]);
  }
  union { v16h h; unsigned u[8]; } r;
#pragma unroll
  for (int i = 0; i < 4; ++i) {
    unsigned xa = (unsigned)__shfl_xor((int)pa[i], 16, 32);
    unsigned xb = (unsigned)__shfl_xor((int)pb[i], 16, 32);
    r.u[i]     = pa[i];
    r.u[4 + i] = pb[i];
  }
  return r.h;
}

static __device__ inline v8f load_bias(const float* bp, int t, int hi) {
  const float* p = bp + t * 16 + hi * 8;
  v8f r;
#pragma unroll
  for (int i = 0; i < 8; ++i) r[i] = p[i];
  return r;
}

static __device__ inline v8f relu8(v8f c) {
#pragma unroll
  for (int i = 0; i < 8; ++i) c[i] = fmaxf(c[i], 0.0f);
  return c;
}

// One 16(out-feat) x 16(batch) output tile: C = bias; C += Wt_tile * in
template <int CH>
static __device__ inline v8f mlp_tile(const _Float16* WP, int fragBase,
                                      const float* bias, int t, int hi,
                                      int lane, const v16h* in) {
  v8f C = load_bias(bias, t, hi);
#pragma unroll
  for (int k = 0; k < CH; ++k) {
    v16h a = *(const v16h*)(WP + (size_t)(fragBase + t * CH + k) * 512 + lane * 16);
    C = __builtin_amdgcn_wmma_f32_16x16x32_f16(false, a, false, in[k],
                                               (short)0, C, false, false);
  }
  return C;
}

// ---------------------------------------------------------------------------
// Main kernel: 512 blocks x 256 threads (8 waves).  Wave w handles batch rows
// [blk*128 + 16w, +16).  Lane (n = lane&15) owns batch column n; hi = lane>=16
// selects the feature sub-rows per the D layout.
// ---------------------------------------------------------------------------
__launch_bounds__(256, 1)
__global__ void odenet_main(const float* __restrict__ x,
                            const float* __restrict__ gb1,
                            const float* __restrict__ gb2,
                            const float* __restrict__ gb3,
                            const float* __restrict__ gWl,
                            const float* __restrict__ gbl,
                            const unsigned short* __restrict__ wp,
                            float* __restrict__ out) {
  extern __shared__ char smem[];
  _Float16* WP = (_Float16*)smem;
  float* B1 = (float*)(smem + LDS_B1);
  float* B2 = (float*)(smem + LDS_B2);
  float* B3 = (float*)(smem + LDS_B3);
  float* WL = (float*)(smem + LDS_WL);
  float* BL = (float*)(smem + LDS_BL);

  int tid = threadIdx.x;

  // ---- stage weights (256KB) + biases + head into LDS -------------------
  {
    uint4*       s = (uint4*)smem;
    const uint4* g = (const uint4*)wp;
    for (int i = tid; i < 16384; i += 256) s[i] = g[i];
    for (int i = tid; i < 256;  i += 256) B1[i] = gb1[i];
    for (int i = tid; i < 256;  i += 256) B2[i] = gb2[i];
    if (tid < 128) B3[tid] = gb3[tid];
    for (int i = tid; i < 1280; i += 256) WL[i] = gWl[i];
    if (tid < 10) BL[tid] = gbl[tid];
  }
  __syncthreads();

  int lane = tid & 31;
  int wave = tid >> 5;
  int hi   = lane >> 4;
  int n    = lane & 15;
  int row  = blockIdx.x * 128 + wave * 16 + n;

  // ---- load x, augment with zeros: yf[ft][v] = state(feat=16ft+v+8hi) ---
  v8f yf[8];
#pragma unroll
  for (int ft = 0; ft < 8; ++ft) {
#pragma unroll
    for (int v = 0; v < 8; ++v) {
      int f = 16 * ft + v + 8 * hi;
      yf[ft][v] = (f < NDATA) ? x[(size_t)row * NDATA + f] : 0.0f;
    }
  }

  const float hstep = 1.0f / NSTEPS;
  const float Ac[4] = {0.5f * hstep, 0.5f * hstep, hstep, 0.0f};
  const float Wc[4] = {1.0f, 2.0f, 2.0f, 1.0f};

  v16h ys[4], h1f[8], h2f[8];
  v8f  acc[8];

#pragma unroll 1
  for (int step = 0; step < NSTEPS; ++step) {
#pragma unroll
    for (int c = 0; c < 4; ++c) ys[c] = packpair(yf[2 * c], yf[2 * c + 1], hi);
#pragma unroll
    for (int t = 0; t < 8; ++t)
#pragma unroll
      for (int i = 0; i < 8; ++i) acc[t][i] = 0.0f;

#pragma unroll 1
    for (int s = 0; s < 4; ++s) {
      // Stop LICM from hoisting the (loop-invariant) LDS weight-fragment
      // loads out of the loops -- that caused them to spill to scratch and
      // starve the WMMAs.  Registers (yf/acc/ys/h1f/h2f) are SSA values and
      // are unaffected by this memory clobber.
      asm volatile("" ::: "memory");

      float As = Ac[s], Ws = Wc[s];

      // layer 1: 128 -> 256  (64 WMMA)
#pragma unroll
      for (int c = 0; c < 8; ++c) {
        v8f t0 = relu8(mlp_tile<4>(WP, 0, B1, 2 * c,     hi, lane, ys));
        v8f t1 = relu8(mlp_tile<4>(WP, 0, B1, 2 * c + 1, hi, lane, ys));
        h1f[c] = packpair(t0, t1, hi);
      }
      // layer 2: 256 -> 256  (128 WMMA)
#pragma unroll
      for (int c = 0; c < 8; ++c) {
        v8f t0 = relu8(mlp_tile<8>(WP, 64, B2, 2 * c,     hi, lane, h1f));
        v8f t1 = relu8(mlp_tile<8>(WP, 64, B2, 2 * c + 1, hi, lane, h1f));
        h2f[c] = packpair(t0, t1, hi);
      }
      // layer 3: 256 -> 128  (64 WMMA) + RK4 stage combine
#pragma unroll
      for (int c = 0; c < 4; ++c) {
        v8f k0 = mlp_tile<8>(WP, 192, B3, 2 * c,     hi, lane, h2f);
        v8f k1 = mlp_tile<8>(WP, 192, B3, 2 * c + 1, hi, lane, h2f);
        acc[2 * c]     += k0 * Ws;
        acc[2 * c + 1] += k1 * Ws;
        if (s < 3) {
          v8f s0 = yf[2 * c]     + k0 * As;
          v8f s1 = yf[2 * c + 1] + k1 * As;
          ys[c] = packpair(s0, s1, hi);
        }
      }
    }

    float g = hstep / 6.0f;
#pragma unroll
    for (int t = 0; t < 8; ++t) yf[t] += acc[t] * g;
  }

  // ---- head: pred = features @ Wl + bl  (partial dot + lane^16 reduce) ---
  float pred[ODIM];
#pragma unroll
  for (int o = 0; o < ODIM; ++o) {
    float p = 0.0f;
#pragma unroll
    for (int ft = 0; ft < 8; ++ft)
#pragma unroll
      for (int v = 0; v < 8; ++v) {
        int f = 16 * ft + v + 8 * hi;
        p += yf[ft][v] * WL[f * ODIM + o];
      }
    p += __shfl_xor(p, 16, 32);
    pred[o] = p + BL[o];
  }
  if (lane < 16) {
#pragma unroll
    for (int o = 0; o < ODIM; ++o) out[(size_t)row * ODIM + o] = pred[o];
  }
}

// ---------------------------------------------------------------------------
extern "C" void kernel_launch(void* const* d_in, const int* in_sizes, int n_in,
                              void* d_out, int out_size, void* d_ws, size_t ws_size,
                              hipStream_t stream) {
  (void)in_sizes; (void)n_in; (void)out_size;
  if (ws_size < 262144) return;

  const float* x  = (const float*)d_in[0];
  const float* W1 = (const float*)d_in[1];
  const float* b1 = (const float*)d_in[2];
  const float* W2 = (const float*)d_in[3];
  const float* b2 = (const float*)d_in[4];
  const float* W3 = (const float*)d_in[5];
  const float* b3 = (const float*)d_in[6];
  const float* Wl = (const float*)d_in[7];
  const float* bl = (const float*)d_in[8];
  unsigned short* wpack = (unsigned short*)d_ws;
  float* out = (float*)d_out;

  hipFuncSetAttribute(reinterpret_cast<const void*>(odenet_main),
                      hipFuncAttributeMaxDynamicSharedMemorySize, LDS_TOTAL);

  prep_pack_weights<<<32, 256, 0, stream>>>(W1, W2, W3, wpack);
  odenet_main<<<512, 256, LDS_TOTAL, stream>>>(x, b1, b2, b3, Wl, bl, wpack, out);
}
